// Actor_56934086475993
// MI455X (gfx1250) — matrix operand order repaired and checked by reading, not compiled, not run on hardware
//
#include <hip/hip_runtime.h>

// ---------------------------------------------------------------------------
// Fused Actor network for MI455X (gfx1250, wave32, WMMA bf16 16x16x32).
// B=16384, V=20, F=15, H=256. One block = 32 batch rows, 8 waves.
// All activations stay in LDS; weights pre-converted to padded bf16 in d_ws.
// Round-1 fix: per-tile GRU accumulation + LDS ping-pong h to kill VGPR spills.
// ---------------------------------------------------------------------------

typedef __attribute__((ext_vector_type(16))) __bf16 v16bf;
typedef __attribute__((ext_vector_type(8)))  float  v8f;

#define BM 32           // batch rows per block
#define NV 20           // vehicles (GRU steps)
#define HH 256          // hidden
#define G3 768          // 3*H
#define KST 32          // padded state feature K (15 -> 32)
#define X0K 288         // padded concat dim (271 -> 288)

__device__ __forceinline__ unsigned short f2bf(float x) {
    union { float f; unsigned u; } c; c.f = x;
    unsigned r = c.u + 0x7FFFu + ((c.u >> 16) & 1u);   // round-to-nearest-even
    return (unsigned short)(r >> 16);
}
__device__ __forceinline__ float bf2f(unsigned short us) {
    union { unsigned u; float f; } c; c.u = ((unsigned)us) << 16;
    return c.f;
}
__device__ __forceinline__ float sigmoidf_(float x) {
    return 1.0f / (1.0f + __expf(-x));
}

union FragBF { uint4 q[2]; v16bf v; };

__device__ __forceinline__ v8f wmma_bf16(v16bf a, v16bf b, v8f c) {
    return __builtin_amdgcn_wmma_f32_16x16x32_bf16(
        /*neg_a=*/false, a, /*neg_b=*/false, b,
        /*c_mod=*/(short)0, c, /*reuse_a=*/false, /*reuse_b=*/false);
}

// A fragment (16x32 bf16), row-major source with leading dim ld (elements).
// Per-lane: m = lane&15, K chunk [koff,koff+16) with koff = (lane>>4)*16.
__device__ __forceinline__ v16bf load_a_frag(const unsigned short* base, int row,
                                             int ld, int k0, int lane) {
    int koff = (lane >> 4) * 16;
    const unsigned short* p = base + row * ld + k0 + koff;
    FragBF f;
    f.q[0] = *reinterpret_cast<const uint4*>(p);
    f.q[1] = *reinterpret_cast<const uint4*>(p + 8);
    return f.v;
}

// B fragment (32x16 bf16) from row-major W[N][K]: lane n = n0+(lane&15),
// K slab = k0 + (lane>=16 ? 16 : 0), 16 contiguous bf16 = 32 contiguous bytes.
__device__ __forceinline__ v16bf load_b_frag(const unsigned short* __restrict__ W,
                                             int n0, int ldk, int k0, int lane) {
    int n  = n0 + (lane & 15);
    int kb = k0 + ((lane >> 4) << 4);
    const uint4* p = reinterpret_cast<const uint4*>(W + (size_t)n * ldk + kb);
    FragBF f; f.q[0] = p[0]; f.q[1] = p[1];
    return f.v;
}

// One dense layer chunk-wise: dst[m][n] = relu(src[m][:] @ W[n][:] + bias[n]).
// src/dst in LDS (bf16), W in global (bf16, row-major [Nout][ldk]).
__device__ __forceinline__ void mlp_layer(const unsigned short* __restrict__ src, int ldk,
                                          const unsigned short* __restrict__ Wg, int Nout,
                                          const float* __restrict__ biasLds,
                                          unsigned short* __restrict__ dst, int ldn,
                                          int wave, int lane) {
    const int mt = wave >> 2, ng = wave & 3;
    const int mrow = mt * 16 + (lane & 15);
    const int hi = (lane >> 4) * 8;
    const v8f vzero = {0.f, 0.f, 0.f, 0.f, 0.f, 0.f, 0.f, 0.f};
    for (int nc = 0; nc < Nout; nc += 256) {
        __builtin_prefetch(Wg + (size_t)nc * ldk, 0, 1);
        v8f acc[4];
        #pragma unroll
        for (int j = 0; j < 4; ++j) acc[j] = vzero;
        #pragma unroll 2
        for (int k0 = 0; k0 < ldk; k0 += 32) {
            v16bf a = load_a_frag(src, mrow, ldk, k0, lane);
            #pragma unroll
            for (int j = 0; j < 4; ++j) {
                v16bf b = load_b_frag(Wg, nc + (ng * 4 + j) * 16, ldk, k0, lane);
                acc[j] = wmma_bf16(a, b, acc[j]);
            }
        }
        #pragma unroll
        for (int j = 0; j < 4; ++j) {
            int n = nc + (ng * 4 + j) * 16 + (lane & 15);
            float bi = biasLds[n];
            #pragma unroll
            for (int e = 0; e < 8; ++e) {
                float v = fmaxf(acc[j][e] + bi, 0.0f);   // ReLU
                int m = mt * 16 + e + hi;
                dst[m * ldn + n] = f2bf(v);
            }
        }
    }
}

// ---------------------------------------------------------------------------
// Weight conversion: f32 -> bf16 with column zero-padding. Run once per matrix.
// ---------------------------------------------------------------------------
__global__ void pad_convert_bf16(const float* __restrict__ src,
                                 unsigned short* __restrict__ dst,
                                 int rows, int scols, int dcols) {
    int i = blockIdx.x * blockDim.x + threadIdx.x;
    int n = rows * dcols;
    if (i >= n) return;
    int r = i / dcols, c = i - r * dcols;
    dst[i] = (c < scols) ? f2bf(src[(size_t)r * scols + c]) : (unsigned short)0;
}

// ---------------------------------------------------------------------------
// Fused Actor kernel
// ---------------------------------------------------------------------------
__global__ __launch_bounds__(256)
void actor_fused_kernel(const float* __restrict__ state,
                        const float* __restrict__ b_ih, const float* __restrict__ b_hh,
                        const float* __restrict__ b1,  const float* __restrict__ b2,
                        const float* __restrict__ b3,  const float* __restrict__ b4,
                        const float* __restrict__ Wp,  const float* __restrict__ bp,
                        const unsigned short* __restrict__ Wih,   // [768][32]
                        const unsigned short* __restrict__ Whh,   // [768][256]
                        const unsigned short* __restrict__ W1p,   // [1024][288]
                        const unsigned short* __restrict__ W2p,   // [1024][1024]
                        const unsigned short* __restrict__ W3p,   // [512][1024]
                        const unsigned short* __restrict__ W4p,   // [256][512]
                        float* __restrict__ out) {
    __shared__ alignas(16) unsigned short bufA[BM * 1024];  // 64 KB: state stage -> y1 -> y3
    __shared__ alignas(16) unsigned short bufB[BM * 1024];  // 64 KB: x0 -> y2 -> y4
    __shared__ alignas(16) unsigned short hPing[BM * HH];   // 16 KB
    __shared__ alignas(16) unsigned short hPong[BM * HH];   // 16 KB
    __shared__ float biasG[G3];     // b_ih+b_hh merged (r,z gates)
    __shared__ float biasIHn[HH];   // b_ih for n gate (outside r*)
    __shared__ float biasBHn[HH];   // b_hh for n gate (inside  r*)
    __shared__ float biasL[1024];   // per-layer MLP bias

    const int tid  = threadIdx.x;
    const int wave = tid >> 5;
    const int lane = tid & 31;
    const int b0   = blockIdx.x * BM;

    const int mt = wave >> 2;        // M tile (0..1)
    const int ng = wave & 3;         // N quarter (0..3)
    const int mrow = mt * 16 + (lane & 15);
    const int hi   = (lane >> 4) * 8;
    const v8f vzero = {0.f, 0.f, 0.f, 0.f, 0.f, 0.f, 0.f, 0.f};

    // ---- stage state tile (bf16, K padded 15->32): bufA[v][m][32] ----
    for (int i = tid; i < NV * BM * KST; i += 256) {
        int v = i >> 10;           // / (32*32)
        int r = i & 1023;
        int m = r >> 5, c = r & 31;
        float val = (c < 15) ? state[(size_t)(b0 + m) * (NV * 15) + v * 15 + c] : 0.0f;
        bufA[i] = f2bf(val);
    }
    // ---- h0 = 0 in hPing ----
    for (int i = tid; i < BM * HH; i += 256) hPing[i] = 0;
    // ---- GRU biases ----
    for (int i = tid; i < G3; i += 256) biasG[i] = b_ih[i] + b_hh[i];
    for (int i = tid; i < HH; i += 256) { biasIHn[i] = b_ih[512 + i]; biasBHn[i] = b_hh[512 + i]; }
    __syncthreads();

    // =================== GRU scan over V=20 steps (h ping-pongs in LDS) ===================
    for (int v = 0; v < NV; ++v) {
        const unsigned short* hs = (v & 1) ? hPong : hPing;   // read side
        unsigned short*       hd = (v & 1) ? hPing : hPong;   // write side
        __builtin_prefetch(Whh, 0, 1);

        // One 16x16 output tile at a time: only 4 v8f accumulators live.
        #pragma unroll 1
        for (int j = 0; j < 4; ++j) {
            const int nt = (ng * 4 + j) * 16;
            const int nl = nt + (lane & 15);
            v8f accR = vzero, accZ = vzero, accHN = vzero, accXN = vzero;

            // input-side pre-activations (K=32, one WMMA per gate)
            v16bf a_s = load_a_frag(bufA + v * (BM * KST), mrow, KST, 0, lane);
            accR  = wmma_bf16(a_s, load_b_frag(Wih,       nt, KST, 0, lane), accR);
            accZ  = wmma_bf16(a_s, load_b_frag(Wih, 256 + nt, KST, 0, lane), accZ);
            accXN = wmma_bf16(a_s, load_b_frag(Wih, 512 + nt, KST, 0, lane), accXN);

            // hidden-side: h @ W_hh^T, K=256, 3 gates per K step
            #pragma unroll 2
            for (int kk = 0; kk < 8; ++kk) {
                v16bf a_h = load_a_frag(hs, mrow, HH, kk * 32, lane);
                accR  = wmma_bf16(a_h, load_b_frag(Whh,       nt, HH, kk * 32, lane), accR);
                accZ  = wmma_bf16(a_h, load_b_frag(Whh, 256 + nt, HH, kk * 32, lane), accZ);
                accHN = wmma_bf16(a_h, load_b_frag(Whh, 512 + nt, HH, kk * 32, lane), accHN);
            }

            // gate math + h update for this tile's (m,n) positions
            float bR = biasG[nl], bZ = biasG[256 + nl];
            float bXN = biasIHn[nl], bHN = biasBHn[nl];
            #pragma unroll
            for (int e = 0; e < 8; ++e) {
                int m = mt * 16 + e + hi;
                float r  = sigmoidf_(accR[e] + bR);
                float z  = sigmoidf_(accZ[e] + bZ);
                float nn = tanhf(accXN[e] + bXN + r * (accHN[e] + bHN));
                float ho = bf2f(hs[m * HH + nl]);
                hd[m * HH + nl] = f2bf((1.0f - z) * nn + z * ho);
            }
        }
        __syncthreads();   // hd complete -> becomes hs of next step
    }
    // after 20 steps the final h is back in hPing

    // =================== build x0 = [state_v0 | h | pad] in bufB ===================
    for (int i = tid; i < BM * X0K; i += 256) {
        int m = i / X0K, c = i - m * X0K;
        unsigned short val;
        if (c < 15)        val = bufA[m * KST + c];            // bf16 state, v=0
        else if (c < 271)  val = hPing[m * HH + (c - 15)];     // final h
        else               val = 0;
        bufB[i] = val;
    }
    __syncthreads();

    // =================== MLP (ping-pong LDS buffers) ===================
    // L1: [32x288] @ [288x1024] -> bufA
    for (int i = tid; i < 1024; i += 256) biasL[i] = b1[i];
    __syncthreads();
    mlp_layer(bufB, X0K, W1p, 1024, biasL, bufA, 1024, wave, lane);
    __syncthreads();
    // L2: [32x1024] @ [1024x1024] -> bufB
    for (int i = tid; i < 1024; i += 256) biasL[i] = b2[i];
    __syncthreads();
    mlp_layer(bufA, 1024, W2p, 1024, biasL, bufB, 1024, wave, lane);
    __syncthreads();
    // L3: [32x1024] @ [1024x512] -> bufA
    for (int i = tid; i < 512; i += 256) biasL[i] = b3[i];
    __syncthreads();
    mlp_layer(bufB, 1024, W3p, 512, biasL, bufA, 512, wave, lane);
    __syncthreads();
    // L4: [32x512] @ [512x256] -> bufB
    for (int i = tid; i < 256; i += 256) biasL[i] = b4[i];
    __syncthreads();
    mlp_layer(bufA, 512, W4p, 256, biasL, bufB, 256, wave, lane);
    __syncthreads();

    // =================== head: tanh(y4 @ Wp^T + bp), one scalar per row ===================
    if (tid < BM) {
        float acc = 0.0f;
        #pragma unroll 8
        for (int k = 0; k < HH; ++k) acc += bf2f(bufB[tid * HH + k]) * Wp[k];
        out[b0 + tid] = tanhf(acc + bp[0]);
    }
}

// ---------------------------------------------------------------------------
// Host launch
// ---------------------------------------------------------------------------
extern "C" void kernel_launch(void* const* d_in, const int* in_sizes, int n_in,
                              void* d_out, int out_size, void* d_ws, size_t ws_size,
                              hipStream_t stream) {
    (void)in_sizes; (void)n_in; (void)out_size; (void)ws_size;
    const float* state = (const float*)d_in[0];
    const float* W_ih  = (const float*)d_in[1];
    const float* W_hh  = (const float*)d_in[2];
    const float* b_ih  = (const float*)d_in[3];
    const float* b_hh  = (const float*)d_in[4];
    const float* W1    = (const float*)d_in[5];
    const float* b1    = (const float*)d_in[6];
    const float* W2    = (const float*)d_in[7];
    const float* b2    = (const float*)d_in[8];
    const float* W3    = (const float*)d_in[9];
    const float* b3    = (const float*)d_in[10];
    const float* W4    = (const float*)d_in[11];
    const float* b4    = (const float*)d_in[12];
    const float* Wp    = (const float*)d_in[13];
    const float* bp    = (const float*)d_in[14];
    float* out = (float*)d_out;

    // bf16 weight arena in workspace (element offsets; all 16B-aligned)
    unsigned short* ws = (unsigned short*)d_ws;
    unsigned short* Wih_p = ws;                       // 768*32    = 24576
    unsigned short* Whh_p = Wih_p + 768 * 32;         // 768*256   = 196608
    unsigned short* W1_p  = Whh_p + 768 * 256;        // 1024*288  = 294912
    unsigned short* W2_p  = W1_p  + 1024 * 288;       // 1024*1024 = 1048576
    unsigned short* W3_p  = W2_p  + 1024 * 1024;      // 512*1024  = 524288
    unsigned short* W4_p  = W3_p  + 512 * 1024;       // 256*512   = 131072

    auto launch_pad = [&](const float* src, unsigned short* dst, int rows, int sc, int dc) {
        int n = rows * dc;
        pad_convert_bf16<<<(n + 255) / 256, 256, 0, stream>>>(src, dst, rows, sc, dc);
    };
    launch_pad(W_ih, Wih_p,  768,   15,   32);
    launch_pad(W_hh, Whh_p,  768,  256,  256);
    launch_pad(W1,   W1_p,  1024,  271,  288);
    launch_pad(W2,   W2_p,  1024, 1024, 1024);
    launch_pad(W3,   W3_p,   512, 1024, 1024);
    launch_pad(W4,   W4_p,   256,  512,  512);

    const int nblocks = 16384 / BM;   // 512
    actor_fused_kernel<<<nblocks, 256, 0, stream>>>(
        state, b_ih, b_hh, b1, b2, b3, b4, Wp, bp,
        Wih_p, Whh_p, W1_p, W2_p, W3_p, W4_p, out);
}
